// SHLightRotateDecorator_14379550507348
// MI455X (gfx1250) — matrix operand order repaired and checked by reading, not compiled, not run on hardware
//
#include <hip/hip_runtime.h>

typedef __attribute__((ext_vector_type(16))) _Float16 v16h;
typedef __attribute__((ext_vector_type(8)))  float    v8f;
typedef __attribute__((ext_vector_type(8)))  int      v8i;

#define PI_A   3.1415926f
#define KTH    (PI_A / 16.0f)      // grid step for th/ph
#define NPIX   512                 // 16 x 32 sphere grid
#define NSAMP  65536
#define WAVES_PER_BLOCK 8          // 256 threads; LDS = 128KB lm + 32KB w = 160KB
#define THREADS_PER_BLOCK (WAVES_PER_BLOCK * 32)
#define SAMP_PER_BLOCK  (WAVES_PER_BLOCK * 16)

union ABFrag { v8i i32; v16h f16; };

// ---------------------------------------------------------------------------
// Prep: wB[n][k] (f16, n-major, n in [0,32), k in [0,512)) and colsum[n] (f32)
//   w[k_coeff, p] = bases[k_coeff, y, x] * sin(th_y) * (pi^2 / 512)
// ---------------------------------------------------------------------------
__global__ void sh_prep_kernel(const float* __restrict__ bases,
                               _Float16* __restrict__ wB,
                               float* __restrict__ colsum) {
    __shared__ float red[512];
    const int n = blockIdx.x;       // 0..31 (>=25 are zero padding)
    const int k = threadIdx.x;      // 0..511 pixel index, p = y*32 + x
    float val = 0.0f;
    if (n < 25) {
        const int y = k >> 5;
        const float th = (y + 0.5f) * KTH;
        val = bases[n * NPIX + k] * __sinf(th) *
              (float)(3.14159265358979323846 * 3.14159265358979323846 / 512.0);
    }
    wB[n * NPIX + k] = (_Float16)val;
    red[k] = val;
    __syncthreads();
    for (int s = 256; s > 0; s >>= 1) {
        if (k < s) red[k] += red[k + s];
        __syncthreads();
    }
    if (k == 0) colsum[n] = red[0];
}

// ---------------------------------------------------------------------------
// Main: per wave, 16 samples. Phase B: lm (f16) into LDS + per-sample energy.
// Phase C: [16 x 512] x [512 x 32] GEMM via v_wmma_f32_16x16x32_f16.
// ---------------------------------------------------------------------------
__global__ void __launch_bounds__(THREADS_PER_BLOCK)
sh_light_rotate_kernel(const int* __restrict__ iternum_p,
                       const _Float16* __restrict__ wB,
                       const float* __restrict__ colsumG,
                       float* __restrict__ out) {
    __shared__ __align__(16) _Float16 lmLds[WAVES_PER_BLOCK][16][NPIX]; // 128 KB
    __shared__ __align__(16) _Float16 wLds[32][NPIX];                   //  32 KB

    const int tid  = threadIdx.x;
    const int wave = tid >> 5;
    const int lane = tid & 31;
    const int m    = lane & 15;     // sample-in-wave / A row / C/D column
    const int h    = lane >> 4;     // lane half

    // pull the (L2-resident) weight matrix toward this WGP early
    __builtin_prefetch(wB + (tid << 6), 0, 3);

    // --- cooperative copy of the weight matrix into LDS (b128 chunks) ------
    {
        const int4* src = (const int4*)wB;
        int4* dst = (int4*)&wLds[0][0];
        #pragma unroll
        for (int i = tid; i < (32 * NPIX * 2) / 16; i += THREADS_PER_BLOCK)
            dst[i] = src[i];
    }

    // --- Phase B: gaussian lobe in log space -------------------------------
    const long long sampleBase = (long long)blockIdx.x * SAMP_PER_BLOCK;
    const long long b = sampleBase + wave * 16 + m;
    const long long nstep = (long long)(*iternum_p) * (long long)NSAMP + b;
    // float64 phase like the reference (f32 range reduction would be wrong)
    const double ph0d = (double)nstep * (2.0 * 3.1415926 / 200.0);
    const float sph0 = (float)sin(ph0d);
    const float cph0 = (float)cos(ph0d);

    float esum = 0.0f;
    for (int yy = 0; yy < 8; ++yy) {
        const int y = h * 8 + yy;                 // lanes m and m+16 split rows
        const float st = __sinf((y + 0.5f) * KTH);
        #pragma unroll 4
        for (int x = 0; x < 32; ++x) {
            float sp, cp;
            __sincosf((x + 0.5f) * KTH, &sp, &cp);
            float d = sph0 * (st * sp) + cph0 * (st * cp);
            d = fminf(1.0f, fmaxf(-1.0f, d));
            const float a = acosf(d);
            const float e = 1e-5f + __expf(-4.0f * a * a);
            esum += e;
            lmLds[wave][m][(y << 5) + x] = (_Float16)__logf(e);
        }
    }
    esum += __shfl_xor(esum, 16);                 // both halves: total energy
    const float cnorm = 7.001f - __logf(esum);    // (+7 - lse + 0.001) folded

    __syncthreads();  // lm visible cross-lane, wLds visible cross-wave

    // --- Phase C: GEMM with WMMA ------------------------------------------
    const int* lmI = (const int*)&lmLds[wave][0][0];  // [16][256] dwords
    const int* wI  = (const int*)&wLds[0][0];         // [32][256] dwords

    v8f acc0 = {0.f, 0.f, 0.f, 0.f, 0.f, 0.f, 0.f, 0.f};
    v8f acc1 = {0.f, 0.f, 0.f, 0.f, 0.f, 0.f, 0.f, 0.f};

    #pragma unroll 4
    for (int kb = 0; kb < NPIX; kb += 32) {
        ABFrag a, b0, b1;
        #pragma unroll
        for (int v = 0; v < 8; ++v) {
            // A 16x32 f16 layout: K = kb + h*16 + v*2
            const int kA = kb + (h << 4) + (v << 1);
            a.i32[v] = lmI[m * 256 + (kA >> 1)];
            // B 32x16 f16 layout: K = kb + h*16 + v*2, column = lane&15 (+16)
            const int kB = kb + (h << 4) + (v << 1);
            b0.i32[v] = wI[m * 256 + (kB >> 1)];
            b1.i32[v] = wI[(m + 16) * 256 + (kB >> 1)];
        }
        acc0 = __builtin_amdgcn_wmma_f32_16x16x32_f16(
            false, a.f16, false, b0.f16, (short)0, acc0, false, false);
        acc1 = __builtin_amdgcn_wmma_f32_16x16x32_f16(
            false, a.f16, false, b1.f16, (short)0, acc1, false, false);
    }

    // --- Epilogue: add c * colsum(w), replicate x3 (RGB) -------------------
    const int n = m;                      // C/D column = lane & 15
    const float cs0 = colsumG[n];
    const float cs1 = colsumG[n + 16];
    #pragma unroll
    for (int r = 0; r < 8; ++r) {
        const int mm = r + (h << 3);      // C/D row: r (lanes 0-15), r+8 (16-31)
        const float cm = __shfl(cnorm, mm);   // lane mm holds sample mm's c
        const long long bb = sampleBase + wave * 16 + mm;
        float* o = out + bb * 75;
        const float v0 = acc0[r] + cm * cs0;
        o[3 * n + 0] = v0; o[3 * n + 1] = v0; o[3 * n + 2] = v0;
        if (n < 9) {                      // coeffs 16..24 live in tile 1
            const float v1 = acc1[r] + cm * cs1;
            o[3 * (16 + n) + 0] = v1;
            o[3 * (16 + n) + 1] = v1;
            o[3 * (16 + n) + 2] = v1;
        }
    }
}

extern "C" void kernel_launch(void* const* d_in, const int* in_sizes, int n_in,
                              void* d_out, int out_size, void* d_ws, size_t ws_size,
                              hipStream_t stream) {
    // inputs: [0] lightid (unused), [1] iternum (scalar int), [2] bases [25*512] f32
    const int*   iternum = (const int*)d_in[1];
    const float* bases   = (const float*)d_in[2];
    float*       out     = (float*)d_out;

    _Float16* wB     = (_Float16*)d_ws;                    // 32*512*2 = 32 KB
    float*    colsum = (float*)((char*)d_ws + 32 * NPIX * 2);

    sh_prep_kernel<<<32, 512, 0, stream>>>(bases, wB, colsum);
    sh_light_rotate_kernel<<<NSAMP / SAMP_PER_BLOCK, THREADS_PER_BLOCK, 0, stream>>>(
        iternum, wB, colsum, out);
}
